// SparseAttention_51256139710612
// MI455X (gfx1250) — matrix operand order repaired and checked by reading, not compiled, not run on hardware
//
#include <hip/hip_runtime.h>
#include <hip/hip_bf16.h>

// NSA-style sparse attention forward for gfx1250 (MI455X).
// All matmuls via v_wmma_f32_16x16x32_bf16 (bf16 in, f32 accum).
// WMMA fragments are 2x16B contiguous vector loads; GEMM is software-pipelined
// with a branch-free modulo-2 schedule (last k-step peeled).

typedef __bf16 bf16;
typedef __attribute__((ext_vector_type(16))) __bf16 bf16x16;
typedef __attribute__((ext_vector_type(8)))  __bf16 bf16x8;
typedef __attribute__((ext_vector_type(8)))  float  f32x8;

#define NEGF (-3.4028234663852886e38f)

__device__ __forceinline__ int clampi(int v, int lo, int hi) {
  return v < lo ? lo : (v > hi ? hi : v);
}

// Low-level fragment loads from a fully-resolved base pointer.
// A-fragment: lane half holds K = base..base+15.
__device__ __forceinline__ bf16x16 frag_a(const bf16* b) {
  bf16x8 lo = *(const bf16x8*)(b);
  bf16x8 h8 = *(const bf16x8*)(b + 8);
  return __builtin_shufflevector(lo, h8, 0, 1, 2, 3, 4, 5, 6, 7, 8, 9, 10, 11, 12,
                                 13, 14, 15);
}
// B-fragment (from N x K transposed source): lane holds K = base..base+15.
__device__ __forceinline__ bf16x16 frag_b(const bf16* b) {
  bf16x8 lo = *(const bf16x8*)(b);
  bf16x8 h8 = *(const bf16x8*)(b + 8);
  return __builtin_shufflevector(lo, h8, 0, 1, 2, 3, 4, 5, 6, 7, 8, 9, 10, 11, 12,
                                 13, 14, 15);
}
// Convenience row/col addressed variants.
__device__ __forceinline__ bf16x16 ld_a(const bf16* p, int ld, int row, int k0,
                                        int hi) {
  return frag_a(p + (size_t)row * ld + k0 + (hi ? 16 : 0));
}
__device__ __forceinline__ bf16x16 ld_bt(const bf16* p, int ld, int row, int k0,
                                         int hi) {
  return frag_b(p + (size_t)row * ld + k0 + (hi ? 16 : 0));
}
__device__ __forceinline__ f32x8 wmma_bf(bf16x16 a, bf16x16 b, f32x8 c) {
  return __builtin_amdgcn_wmma_f32_16x16x32_bf16(false, a, false, b, (short)0, c,
                                                 false, false);
}

// ---------------- elementwise / prep kernels ----------------

// dst[Npad][K] = transpose(src[K][N]), zero-padding rows N..Npad-1
__global__ void convert_t_kernel(const float* src, bf16* dst, int K, int N,
                                 int Npad) {
  int idx = blockIdx.x * blockDim.x + threadIdx.x;
  if (idx >= Npad * K) return;
  int n = idx / K, k = idx % K;
  dst[idx] = (n < N) ? (bf16)src[k * N + n] : (bf16)0.f;
}

__global__ void rmsnorm_kernel(const float* inp, const float* nw, bf16* xn) {
  int s = blockIdx.x, tid = threadIdx.x;
  __shared__ float red[256];
  float a = 0.f;
  for (int c = tid; c < 512; c += 256) {
    float v = inp[s * 512 + c];
    a += v * v;
  }
  red[tid] = a;
  __syncthreads();
  for (int o = 128; o > 0; o >>= 1) {
    if (tid < o) red[tid] += red[tid + o];
    __syncthreads();
  }
  float r = rsqrtf(red[0] / 512.f + 1.1920929e-7f);
  for (int c = tid; c < 512; c += 256)
    xn[s * 512 + c] = (bf16)(inp[s * 512 + c] * r * nw[c]);
}

// split qkv, apply interleaved RoPE, build compress-MLP block inputs,
// and store V transposed [kh][dh][S] for vectorized PV fragments.
__global__ void split_rope_kernel(const float* qkv, const float* k_pos,
                                  const float* v_pos, bf16* q_bf, bf16* qr_bf,
                                  bf16* kr_bf, bf16* vT_bf, bf16* kb_bf,
                                  bf16* vb_bf) {
  int s = blockIdx.x, tid = threadIdx.x;
  int nc = s >> 5, cb = s & 31;
  for (int col = tid; col < 768; col += blockDim.x) {
    float v  = qkv[s * 768 + col];
    float vp = qkv[s * 768 + (col ^ 1)];
    int d = col & 63;
    float fi = (float)(d >> 1);
    float theta = (float)s * __powf(10000.f, -2.f * fi / 64.f);
    float cs = __cosf(theta), sn = __sinf(theta);
    float rot = (d & 1) ? vp : -vp;
    float vr = v * cs + rot * sn;
    if (col < 512) {
      int h = col >> 6;
      q_bf[(h * 2048 + s) * 64 + d]  = (bf16)v;
      qr_bf[(h * 2048 + s) * 64 + d] = (bf16)vr;
    } else if (col < 640) {
      int kh = (col - 512) >> 6;
      kr_bf[(kh * 2048 + s) * 64 + d] = (bf16)vr;
      kb_bf[(kh * 64 + nc) * 2048 + cb * 64 + d] =
          (bf16)(v + k_pos[(kh * 32 + cb) * 64 + d]);
    } else {
      int kh = (col - 640) >> 6;
      vT_bf[(size_t)(kh * 64 + d) * 2048 + s] = (bf16)v;
      vb_bf[(kh * 64 + nc) * 2048 + cb * 64 + d] =
          (bf16)(v + v_pos[(kh * 32 + cb) * 64 + d]);
    }
  }
}

// compressed K: [kh][80 keys][64] (row0 = mem_kv, 1..64 = MLP out, pad 0)
// compressed V: transposed [kh][64][96]
__global__ void ckcv_build_kernel(const float* ckc, const float* cvc,
                                  const float* mem_kv, bf16* ck_bf, bf16* cvT_bf) {
  int idx = blockIdx.x * blockDim.x + threadIdx.x;
  if (idx < 2 * 80 * 64) {
    int d = idx & 63, j = (idx >> 6) % 80, kh = idx / (80 * 64);
    float v;
    if (j == 0) v = mem_kv[(0 * 2 + kh) * 64 + d];
    else if (j <= 64) v = ckc[(kh * 64 + (j - 1)) * 64 + d];
    else v = 0.f;
    ck_bf[(kh * 80 + j) * 64 + d] = (bf16)v;
  }
  int i2 = idx - 2 * 80 * 64;
  if (i2 >= 0 && i2 < 2 * 96 * 64) {
    int j = i2 % 96, d = (i2 / 96) & 63, kh = i2 / (96 * 64);
    float v;
    if (j == 0) v = mem_kv[(1 * 2 + kh) * 64 + d];
    else if (j <= 64) v = cvc[(kh * 64 + (j - 1)) * 64 + d];
    else v = 0.f;
    cvT_bf[(kh * 64 + d) * 96 + j] = (bf16)v;
  }
}

// ---------------- generic bf16 WMMA GEMM (software-pipelined) ----------------
// C[M,Npad] = act(A[M,K] * BT[Npad,K]^T + bias[N]); one 16x64 tile per wave.
// Branch-free modulo-2 pipeline: body loads step k+1 fragments unconditionally
// while issuing WMMAs for step k; the final k-step is peeled after the loop.
__global__ void gemm_bf16_kernel(const bf16* A, const bf16* BT, const float* bias,
                                 float* Cf, bf16* Cbf, int M, int N, int Npad,
                                 int K, int act) {
  int wave = (blockIdx.x * blockDim.x + threadIdx.x) >> 5;
  int lane = threadIdx.x & 31;
  int tn = Npad >> 6, tm = M >> 4;
  if (wave >= tm * tn) return;
  int ti = wave / tn, tj = wave % tn;
  int m0 = ti * 16, n0 = tj * 64;
  int hi = lane >> 4, ln = lane & 15;
  int ho8 = hi ? 8 : 0, ho16 = hi ? 16 : 0;

  const bf16* Arow  = A + (size_t)(m0 + ln) * K + ho16;
  const bf16* Brow0 = BT + (size_t)(n0 + 0 * 16 + ln) * K + ho16;
  const bf16* Brow1 = BT + (size_t)(n0 + 1 * 16 + ln) * K + ho16;
  const bf16* Brow2 = BT + (size_t)(n0 + 2 * 16 + ln) * K + ho16;
  const bf16* Brow3 = BT + (size_t)(n0 + 3 * 16 + ln) * K + ho16;

  f32x8 c0 = {0.f, 0.f, 0.f, 0.f, 0.f, 0.f, 0.f, 0.f};
  f32x8 c1 = c0, c2 = c0, c3 = c0;

  bf16x16 a  = frag_a(Arow);
  bf16x16 b0 = frag_b(Brow0), b1 = frag_b(Brow1);
  bf16x16 b2 = frag_b(Brow2), b3 = frag_b(Brow3);

  int k0 = 0;
#pragma unroll 2
  for (; k0 + 32 < K; k0 += 32) {
    int kn = k0 + 32;
    bf16x16 an  = frag_a(Arow + kn);
    bf16x16 b0n = frag_b(Brow0 + kn);
    bf16x16 b1n = frag_b(Brow1 + kn);
    bf16x16 b2n = frag_b(Brow2 + kn);
    bf16x16 b3n = frag_b(Brow3 + kn);
    __builtin_prefetch(Arow + kn + 64, 0, 1);
    __builtin_prefetch(Brow1 + kn + 64, 0, 1);
    c0 = wmma_bf(a, b0, c0);
    c1 = wmma_bf(a, b1, c1);
    c2 = wmma_bf(a, b2, c2);
    c3 = wmma_bf(a, b3, c3);
    a = an;
    b0 = b0n; b1 = b1n; b2 = b2n; b3 = b3n;
  }
  // peeled final k-step
  c0 = wmma_bf(a, b0, c0);
  c1 = wmma_bf(a, b1, c1);
  c2 = wmma_bf(a, b2, c2);
  c3 = wmma_bf(a, b3, c3);

  f32x8 acc[4] = {c0, c1, c2, c3};
#pragma unroll
  for (int q = 0; q < 4; ++q) {
    int n = n0 + q * 16 + ln;
#pragma unroll
    for (int r = 0; r < 8; ++r) {
      int m = m0 + r + ho8;
      float v = acc[q][r];
      if (bias && n < N) v += bias[n];
      if (act == 1) v = v > 0.f ? v : 0.f;
      else if (act == 2) v = 1.f / (1.f + __expf(-v));
      if (Cf)  Cf[(size_t)m * Npad + n] = v;
      if (Cbf) Cbf[(size_t)m * Npad + n] = (bf16)v;
    }
  }
}

// ---------------- compressed attention (unrotated q/k) ----------------
__global__ void cattn_kernel(const bf16* q_bf, const bf16* ck_bf, const bf16* cvT_bf,
                             float* csim_g, float* c_out) {
  int bid = blockIdx.x;
  int st = bid & 127, g = (bid >> 7) & 3, kh = bid >> 9;
  int s0 = st * 16;
  int lane = threadIdx.x, hi = lane >> 4, ln = lane & 15;
  __shared__ __align__(16) float sm[16][80];
  __shared__ __align__(16) bf16 at[16][96];
  const bf16* Q   = q_bf + (size_t)((kh * 4 + g) * 2048) * 64;
  const bf16* CK  = ck_bf + (size_t)kh * 80 * 64;
  const bf16* CVT = cvT_bf + (size_t)kh * 64 * 96;
  bf16x16 a0 = ld_a(Q, 64, s0 + ln, 0, hi);
  bf16x16 a1 = ld_a(Q, 64, s0 + ln, 32, hi);
  for (int t = 0; t < 5; ++t) {
    int n0 = t * 16;
    f32x8 c = {0.f, 0.f, 0.f, 0.f, 0.f, 0.f, 0.f, 0.f};
    c = wmma_bf(a0, ld_bt(CK, 64, n0 + ln, 0, hi), c);
    c = wmma_bf(a1, ld_bt(CK, 64, n0 + ln, 32, hi), c);
    int n = n0 + ln;
    int ckpos = (n == 0) ? -1 : n * 32 - 1;
#pragma unroll
    for (int r = 0; r < 8; ++r) {
      int m = r + (hi ? 8 : 0);
      int s = s0 + m;
      float v = (n < 65 && ckpos < s) ? c[r] * 0.125f : NEGF;
      sm[m][n] = v;
      if (n >= 1 && n < 65)
        csim_g[(size_t)((kh * 4 + g) * 2048 + s) * 64 + (n - 1)] = v;
    }
  }
  __syncthreads();
  if (lane < 16) {
    int row = lane;
    float mx = NEGF;
    for (int n = 0; n < 80; ++n) mx = fmaxf(mx, sm[row][n]);
    float sum = 0.f;
    for (int n = 0; n < 80; ++n) sum += __expf(sm[row][n] - mx);
    float inv = 1.f / sum;
    for (int n = 0; n < 96; ++n)
      at[row][n] = (n < 80) ? (bf16)(__expf(sm[row][n] - mx) * inv) : (bf16)0.f;
  }
  __syncthreads();
  bf16x16 p0 = ld_a(&at[0][0], 96, ln, 0, hi);
  bf16x16 p1 = ld_a(&at[0][0], 96, ln, 32, hi);
  bf16x16 p2 = ld_a(&at[0][0], 96, ln, 64, hi);
  for (int t = 0; t < 4; ++t) {
    int n0 = t * 16;
    f32x8 c = {0.f, 0.f, 0.f, 0.f, 0.f, 0.f, 0.f, 0.f};
    c = wmma_bf(p0, ld_bt(CVT, 96, n0 + ln, 0, hi), c);
    c = wmma_bf(p1, ld_bt(CVT, 96, n0 + ln, 32, hi), c);
    c = wmma_bf(p2, ld_bt(CVT, 96, n0 + ln, 64, hi), c);
#pragma unroll
    for (int r = 0; r < 8; ++r) {
      int m = r + (hi ? 8 : 0);
      c_out[(size_t)((kh * 4 + g) * 2048 + s0 + m) * 64 + n0 + ln] = c[r];
    }
  }
}

// ---------------- importance softmax + top-k selection ----------------
__global__ void topk_kernel(const float* csim_g, int* sel_idx, int* sel_ok) {
  int idx = blockIdx.x * blockDim.x + threadIdx.x;
  if (idx >= 2 * 2048) return;
  int kh = idx >> 11, s = idx & 2047;
  float imp[64];
  for (int j = 0; j < 64; ++j) {
    float a = 0.f;
    for (int g = 0; g < 4; ++g)
      a += csim_g[(size_t)((kh * 4 + g) * 2048 + s) * 64 + j];
    imp[j] = a * 0.25f;
  }
  float mx = -1e3f;  // leading pad value
  for (int j = 0; j < 64; ++j) mx = fmaxf(mx, imp[j]);
  float sum = __expf(-1e3f - mx);
  for (int j = 0; j < 64; ++j) {
    imp[j] = __expf(imp[j] - mx);
    sum += imp[j];
  }
  float inv = 1.f / sum;
  for (int j = 0; j < 64; ++j) imp[j] *= inv;
  for (int t = 0; t < 8; ++t) {
    int best = 0;
    float bv = -1.f;
    for (int j = 0; j < 64; ++j)
      if (imp[j] > bv) { bv = imp[j]; best = j; }
    sel_idx[(kh * 2048 + s) * 9 + t] = best;
    sel_ok[(kh * 2048 + s) * 8 + t] = (bv > 1e-10f) ? 1 : 0;
    imp[best] = -2.f;
  }
  sel_idx[(kh * 2048 + s) * 9 + 8] = s >> 5;  // own block appended
}

// ---------------- fine (selected-block) attention ----------------
// One wave per (kh, s); 4 grouped queries share the gather; WMMA rows carry g=m&3.
__global__ void fattn_kernel(const bf16* qr, const bf16* kr, const bf16* vT,
                             const int* sel_idx, const int* sel_ok, float* f_out) {
  int bid = blockIdx.x;
  int s = bid & 2047, kh = bid >> 11;
  int lane = threadIdx.x, hi = lane >> 4, ln = lane & 15;
  __shared__ __align__(16) float sm[4][288];
  __shared__ __align__(16) bf16 at[4][288];
  __shared__ int sel[9];
  __shared__ int ok[9];
  if (lane < 9) {
    sel[lane] = sel_idx[(kh * 2048 + s) * 9 + lane];
    ok[lane] = (lane < 8) ? sel_ok[(kh * 2048 + s) * 8 + lane] : 1;
  }
  __syncthreads();
  const bf16* Kk = kr + (size_t)kh * 2048 * 64;
  const bf16* VT = vT + (size_t)kh * 64 * 2048;
  int gm = ln & 3;
  const bf16* Qrow = qr + (size_t)((kh * 4 + gm) * 2048 + s) * 64;
  bf16x16 a0 = ld_a(Qrow, 64, 0, 0, hi);
  bf16x16 a1 = ld_a(Qrow, 64, 0, 32, hi);
  for (int t = 0; t < 18; ++t) {
    int n0 = t * 16;
    int jj = n0 + ln, blk = jj >> 5, rw = jj & 31;
    int srow = sel[blk] * 32 + rw;
    f32x8 c = {0.f, 0.f, 0.f, 0.f, 0.f, 0.f, 0.f, 0.f};
    c = wmma_bf(a0, ld_bt(Kk, 64, srow, 0, hi), c);
    c = wmma_bf(a1, ld_bt(Kk, 64, srow, 32, hi), c);
    bool valid = (blk < 8) ? (ok[blk] != 0) : (rw <= (s & 31));
#pragma unroll
    for (int r = 0; r < 8; ++r) {
      int m = r + (hi ? 8 : 0);
      if (m < 4) sm[m][jj] = valid ? c[r] * 0.125f : NEGF;
    }
  }
  __syncthreads();
  if (lane < 4) {
    float mx = NEGF;
    for (int n = 0; n < 288; ++n) mx = fmaxf(mx, sm[lane][n]);
    float sum = 0.f;
    for (int n = 0; n < 288; ++n) sum += __expf(sm[lane][n] - mx);
    float inv = 1.f / sum;
    for (int n = 0; n < 288; ++n)
      at[lane][n] = (bf16)(__expf(sm[lane][n] - mx) * inv);
  }
  __syncthreads();
  for (int t = 0; t < 4; ++t) {
    int n0 = t * 16;
    f32x8 c = {0.f, 0.f, 0.f, 0.f, 0.f, 0.f, 0.f, 0.f};
    for (int u = 0; u < 9; ++u) {
      // A: attention probs row gm, K-chunk u; B: V^T rows = dh, keys of block u
      bf16x16 a = ld_a(&at[gm][0], 288, 0, u * 32, hi);
      bf16x16 b = ld_bt(VT, 2048, n0 + ln, sel[u] * 32, hi);
      c = wmma_bf(a, b, c);
    }
#pragma unroll
    for (int r = 0; r < 8; ++r) {
      int m = r + (hi ? 8 : 0);
      if (m < 4)
        f_out[(size_t)((kh * 4 + m) * 2048 + s) * 64 + n0 + ln] = c[r];
    }
  }
}

// ---------------- sliding-window attention ----------------
// One wave per (kh, g, 16-query tile); 144-key band padded to 160.
__global__ void sattn_kernel(const bf16* qr, const bf16* kr, const bf16* vT,
                             float* s_out) {
  int bid = blockIdx.x;
  int st = bid & 127, g = (bid >> 7) & 3, kh = bid >> 9;
  int s0 = st * 16;
  int lane = threadIdx.x, hi = lane >> 4, ln = lane & 15;
  __shared__ __align__(16) float sm[16][160];
  __shared__ __align__(16) bf16 at[16][160];
  const bf16* Q  = qr + (size_t)((kh * 4 + g) * 2048) * 64;
  const bf16* Kk = kr + (size_t)kh * 2048 * 64;
  const bf16* VT = vT + (size_t)kh * 64 * 2048;
  int kb0 = s0 - 128;
  bf16x16 a0 = ld_a(Q, 64, s0 + ln, 0, hi);
  bf16x16 a1 = ld_a(Q, 64, s0 + ln, 32, hi);
  for (int t = 0; t < 10; ++t) {
    int n0 = t * 16;
    int p = kb0 + n0 + ln;
    int pc = clampi(p, 0, 2047);
    f32x8 c = {0.f, 0.f, 0.f, 0.f, 0.f, 0.f, 0.f, 0.f};
    c = wmma_bf(a0, ld_bt(Kk, 64, pc, 0, hi), c);
    c = wmma_bf(a1, ld_bt(Kk, 64, pc, 32, hi), c);
#pragma unroll
    for (int r = 0; r < 8; ++r) {
      int m = r + (hi ? 8 : 0);
      int i = s0 + m;
      int diff = i - p;
      sm[m][n0 + ln] = (p >= 0 && diff >= 0 && diff <= 128) ? c[r] * 0.125f : NEGF;
    }
  }
  __syncthreads();
  if (lane < 16) {
    float mx = NEGF;
    for (int n = 0; n < 160; ++n) mx = fmaxf(mx, sm[lane][n]);
    float sum = 0.f;
    for (int n = 0; n < 160; ++n) sum += __expf(sm[lane][n] - mx);
    float inv = 1.f / sum;
    for (int n = 0; n < 160; ++n)
      at[lane][n] = (bf16)(__expf(sm[lane][n] - mx) * inv);
  }
  __syncthreads();
  bf16x16 pf[5];
#pragma unroll
  for (int u = 0; u < 5; ++u) pf[u] = ld_a(&at[0][0], 160, ln, u * 32, hi);
  for (int t = 0; t < 4; ++t) {
    int n0 = t * 16;
    f32x8 c = {0.f, 0.f, 0.f, 0.f, 0.f, 0.f, 0.f, 0.f};
    for (int u = 0; u < 5; ++u) {
      // key chunk base, clamped to a valid window (out-of-range cols have attn=0)
      int k0c = clampi(kb0 + u * 32, 0, 2048 - 32);
      bf16x16 b = ld_bt(VT, 2048, n0 + ln, k0c, hi);
      c = wmma_bf(pf[u], b, c);
    }
#pragma unroll
    for (int r = 0; r < 8; ++r) {
      int m = r + (hi ? 8 : 0);
      s_out[(size_t)((kh * 4 + g) * 2048 + s0 + m) * 64 + n0 + ln] = c[r];
    }
  }
}

// ---------------- gated combine ----------------
__global__ void combine_kernel(const float* gates, const float* c_out,
                               const float* f_out, const float* s_out, bf16* oh) {
  int s = blockIdx.x, tid = threadIdx.x;  // 512 threads: h*64+d
  int h = tid >> 6, d = tid & 63;
  float g0 = gates[s * 64 + h * 3 + 0];
  float g1 = gates[s * 64 + h * 3 + 1];
  float g2 = gates[s * 64 + h * 3 + 2];
  size_t idx = (size_t)(h * 2048 + s) * 64 + d;
  oh[s * 512 + tid] = (bf16)(g0 * c_out[idx] + g1 * f_out[idx] + g2 * s_out[idx]);
}

// ---------------- host launch ----------------
static void launch_gemm(const bf16* A, const bf16* BT, const float* bias, float* Cf,
                        bf16* Cbf, int M, int N, int Npad, int K, int act,
                        hipStream_t stream) {
  int waves = (M / 16) * (Npad / 64);
  int threads = 128;
  int blocks = (waves * 32 + threads - 1) / threads;
  gemm_bf16_kernel<<<blocks, threads, 0, stream>>>(A, BT, bias, Cf, Cbf, M, N, Npad,
                                                   K, act);
}

extern "C" void kernel_launch(void* const* d_in, const int* in_sizes, int n_in,
                              void* d_out, int out_size, void* d_ws, size_t ws_size,
                              hipStream_t stream) {
  (void)in_sizes; (void)n_in; (void)out_size;
  const float* inp    = (const float*)d_in[0];
  const float* norm_w = (const float*)d_in[1];
  const float* w_qkv  = (const float*)d_in[2];
  const float* mem_kv = (const float*)d_in[3];
  const float* k_pos  = (const float*)d_in[4];
  const float* v_pos  = (const float*)d_in[5];
  const float* kc_w1  = (const float*)d_in[6];
  const float* kc_b1  = (const float*)d_in[7];
  const float* kc_w2  = (const float*)d_in[8];
  const float* kc_b2  = (const float*)d_in[9];
  const float* vc_w1  = (const float*)d_in[10];
  const float* vc_b1  = (const float*)d_in[11];
  const float* vc_w2  = (const float*)d_in[12];
  const float* vc_b2  = (const float*)d_in[13];
  const float* comb_w = (const float*)d_in[14];
  const float* comb_b = (const float*)d_in[15];
  const float* out_w  = (const float*)d_in[16];
  float* outp = (float*)d_out;

  size_t off = 0;
  auto alloc = [&](size_t bytes) -> void* {
    off = (off + 255) & ~(size_t)255;
    void* p = (char*)d_ws + off;
    off += bytes;
    return p;
  };
  bf16* xn_bf    = (bf16*)alloc((size_t)2048 * 512 * 2);
  bf16* wqkvT    = (bf16*)alloc((size_t)768 * 512 * 2);
  float* qkv_f   = (float*)alloc((size_t)2048 * 768 * 4);
  bf16* q_bf     = (bf16*)alloc((size_t)8 * 2048 * 64 * 2);
  bf16* qr_bf    = (bf16*)alloc((size_t)8 * 2048 * 64 * 2);
  bf16* kr_bf    = (bf16*)alloc((size_t)2 * 2048 * 64 * 2);
  bf16* vT_bf    = (bf16*)alloc((size_t)2 * 64 * 2048 * 2);
  bf16* kb_bf    = (bf16*)alloc((size_t)128 * 2048 * 2);
  bf16* vb_bf    = (bf16*)alloc((size_t)128 * 2048 * 2);
  bf16* kcw1T    = (bf16*)alloc((size_t)2048 * 2048 * 2);
  bf16* kcw2T    = (bf16*)alloc((size_t)64 * 2048 * 2);
  bf16* vcw1T    = (bf16*)alloc((size_t)2048 * 2048 * 2);
  bf16* vcw2T    = (bf16*)alloc((size_t)64 * 2048 * 2);
  bf16* combT    = (bf16*)alloc((size_t)64 * 512 * 2);
  bf16* outwT    = (bf16*)alloc((size_t)512 * 512 * 2);
  bf16* hidk_bf  = (bf16*)alloc((size_t)128 * 2048 * 2);
  bf16* hidv_bf  = (bf16*)alloc((size_t)128 * 2048 * 2);
  float* ckc_f   = (float*)alloc((size_t)128 * 64 * 4);
  float* cvc_f   = (float*)alloc((size_t)128 * 64 * 4);
  bf16* ck_bf    = (bf16*)alloc((size_t)2 * 80 * 64 * 2);
  bf16* cvT_bf   = (bf16*)alloc((size_t)2 * 64 * 96 * 2);
  float* csim_g  = (float*)alloc((size_t)8 * 2048 * 64 * 4);
  float* c_out   = (float*)alloc((size_t)8 * 2048 * 64 * 4);
  float* f_out   = (float*)alloc((size_t)8 * 2048 * 64 * 4);
  float* s_out   = (float*)alloc((size_t)8 * 2048 * 64 * 4);
  float* gates_f = (float*)alloc((size_t)2048 * 64 * 4);
  bf16* oh_bf    = (bf16*)alloc((size_t)2048 * 512 * 2);
  int* sel_idx   = (int*)alloc((size_t)2 * 2048 * 9 * 4);
  int* sel_ok    = (int*)alloc((size_t)2 * 2048 * 8 * 4);
  if (off > ws_size) return;  // scratch too small: no-op

  auto cvt = [&](const float* src, bf16* dst, int K, int N, int Npad) {
    int n = Npad * K;
    convert_t_kernel<<<(n + 255) / 256, 256, 0, stream>>>(src, dst, K, N, Npad);
  };
  cvt(w_qkv, wqkvT, 512, 768, 768);
  cvt(kc_w1, kcw1T, 2048, 2048, 2048);
  cvt(kc_w2, kcw2T, 2048, 64, 64);
  cvt(vc_w1, vcw1T, 2048, 2048, 2048);
  cvt(vc_w2, vcw2T, 2048, 64, 64);
  cvt(comb_w, combT, 512, 24, 64);
  cvt(out_w, outwT, 512, 512, 512);

  rmsnorm_kernel<<<2048, 256, 0, stream>>>(inp, norm_w, xn_bf);

  // qkv = xn @ w_qkv
  launch_gemm(xn_bf, wqkvT, nullptr, qkv_f, nullptr, 2048, 768, 768, 512, 0, stream);

  split_rope_kernel<<<2048, 256, 0, stream>>>(qkv_f, k_pos, v_pos, q_bf, qr_bf,
                                              kr_bf, vT_bf, kb_bf, vb_bf);

  // compress MLPs (bias + ReLU fused, bf16 hidden)
  launch_gemm(kb_bf, kcw1T, kc_b1, nullptr, hidk_bf, 128, 2048, 2048, 2048, 1, stream);
  launch_gemm(hidk_bf, kcw2T, kc_b2, ckc_f, nullptr, 128, 64, 64, 2048, 0, stream);
  launch_gemm(vb_bf, vcw1T, vc_b1, nullptr, hidv_bf, 128, 2048, 2048, 2048, 1, stream);
  launch_gemm(hidv_bf, vcw2T, vc_b2, cvc_f, nullptr, 128, 64, 64, 2048, 0, stream);

  ckcv_build_kernel<<<(2 * 80 * 64 + 2 * 96 * 64 + 255) / 256, 256, 0, stream>>>(
      ckc_f, cvc_f, mem_kv, ck_bf, cvT_bf);

  cattn_kernel<<<1024, 32, 0, stream>>>(q_bf, ck_bf, cvT_bf, csim_g, c_out);
  topk_kernel<<<16, 256, 0, stream>>>(csim_g, sel_idx, sel_ok);
  fattn_kernel<<<4096, 32, 0, stream>>>(qr_bf, kr_bf, vT_bf, sel_idx, sel_ok, f_out);
  sattn_kernel<<<1024, 32, 0, stream>>>(qr_bf, kr_bf, vT_bf, s_out);

  // gates = sigmoid(xn @ comb_w + comb_b)
  launch_gemm(xn_bf, combT, comb_b, gates_f, nullptr, 2048, 24, 64, 512, 2, stream);
  combine_kernel<<<2048, 512, 0, stream>>>(gates_f, c_out, f_out, s_out, oh_bf);

  // out = oh @ out_w
  launch_gemm(oh_bf, outwT, nullptr, outp, nullptr, 2048, 512, 512, 512, 0, stream);
}
